// MSID_6451040879214
// MI455X (gfx1250) — compile-verified, matrix operand order checked by reading, not run on hardware
//
#include <hip/hip_runtime.h>
#include <cstddef>
#include <cstdint>

// ---------------- problem constants (from reference) ----------------
#define N_PTS 8192
#define DIMS  256
#define NV    100
#define KNBR  5      // k=5 neighbors (self excluded during selection)
#define MLZ   10     // Lanczos steps
#define NTSC  256    // number of t samples
#define NREO  3      // reortho passes

// ---------------- CDNA5 WMMA types ----------------
typedef __attribute__((ext_vector_type(8)))  float  v8f;
typedef __attribute__((ext_vector_type(16))) __bf16 v16bf;

union BF16Frag { v16bf v; unsigned int u[8]; };

__device__ __forceinline__ unsigned short f32_to_bf16_rne(float f) {
  unsigned int u = __float_as_uint(f);
  u = u + 0x7FFFu + ((u >> 16) & 1u);
  return (unsigned short)(u >> 16);
}
__device__ __forceinline__ float bf16_bits_to_f32(unsigned short h) {
  return __uint_as_float(((unsigned int)h) << 16);
}

__device__ __forceinline__ void load_frag(BF16Frag& f, const unsigned short* p,
                                          int k0, int k1) {
  uint4 t0 = *(const uint4*)(p + k0);   // 8 contiguous bf16 (16B)
  uint4 t1 = *(const uint4*)(p + k1);   // 8 contiguous bf16 (16B)
  f.u[0] = t0.x; f.u[1] = t0.y; f.u[2] = t0.z; f.u[3] = t0.w;
  f.u[4] = t1.x; f.u[5] = t1.y; f.u[6] = t1.z; f.u[7] = t1.w;
}

// ---------------- stage 1: row norms + bf16 hi/lo split ----------------
__global__ void k_rownorm_cvt(const float* __restrict__ x,
                              unsigned short* __restrict__ hi,
                              unsigned short* __restrict__ lo,
                              float* __restrict__ dd) {
  const int row = blockIdx.x;
  const int c   = threadIdx.x;              // blockDim == DIMS == 256
  const float v = x[(size_t)row * DIMS + c];
  const unsigned short hb = f32_to_bf16_rne(v);
  const float r = v - bf16_bits_to_f32(hb);
  const unsigned short lb = f32_to_bf16_rne(r);
  hi[(size_t)row * DIMS + c] = hb;
  lo[(size_t)row * DIMS + c] = lb;
  __shared__ float red[DIMS];
  red[c] = v * v;
  __syncthreads();
  for (int s = DIMS / 2; s > 0; s >>= 1) {
    if (c < s) red[c] += red[c + s];
    __syncthreads();
  }
  if (c == 0) dd[row] = red[0];
}

// ---------------- stage 2: fused bf16x3 WMMA Gram + streaming top-5 --------
// block = 256 threads (8 waves). 32 rows/block, 64 cols/iter (2x4 wave grid).
__global__ void __launch_bounds__(256)
k_dist_topk(const unsigned short* __restrict__ xhi,
            const unsigned short* __restrict__ xlo,
            const float* __restrict__ dd,
            int* __restrict__ knn) {
  __shared__ __align__(16) unsigned short sAhi[32][DIMS];   // 16 KB
  __shared__ __align__(16) unsigned short sAlo[32][DIMS];   // 16 KB
  __shared__ float sDist[32][64];                           // 8 KB

  const int tid  = threadIdx.x;
  const int lane = tid & 31;
  const int wave = tid >> 5;
  const int rs   = wave & 1;     // row strip (2 x 16 rows)
  const int cs   = wave >> 1;    // col strip (4 x 16 cols)
  const int rowBase = blockIdx.x * 32;
  const int mrow = lane & 15;
  const int half = lane >> 4;

  for (int idx = tid; idx < 32 * DIMS; idx += 256) {
    const int r = idx >> 8, c = idx & 255;
    sAhi[r][c] = xhi[(size_t)(rowBase + r) * DIMS + c];
    sAlo[r][c] = xlo[(size_t)(rowBase + r) * DIMS + c];
  }
  __syncthreads();

  float bestV[KNBR];
  int   bestI[KNBR];
  float worst = 3.4e38f;
  int   worstSlot = 0;
  for (int e = 0; e < KNBR; ++e) { bestV[e] = 3.4e38f; bestI[e] = 0; }

  const unsigned short* ah = &sAhi[rs * 16 + mrow][0];
  const unsigned short* al = &sAlo[rs * 16 + mrow][0];

  for (int cbase = 0; cbase < N_PTS; cbase += 64) {
    const int gcol = cbase + cs * 16 + mrow;
    const unsigned short* bh = xhi + (size_t)gcol * DIMS;
    const unsigned short* bl = xlo + (size_t)gcol * DIMS;

    v8f acc = {0.f, 0.f, 0.f, 0.f, 0.f, 0.f, 0.f, 0.f};

#pragma unroll
    for (int kc = 0; kc < DIMS; kc += 32) {
      const int k0 = kc + half * 8;        // K pairs for this lane-half
      const int k1 = kc + 16 + half * 8;
      BF16Frag aH, aL, bH, bL;
      load_frag(aH, ah, k0, k1);
      load_frag(aL, al, k0, k1);
      load_frag(bH, bh, k0, k1);
      load_frag(bL, bl, k0, k1);
      // S = hi*hi + hi*lo + lo*hi  (bf16x3 ~ fp32 accuracy for ranking)
      acc = __builtin_amdgcn_wmma_f32_16x16x32_bf16(false, aH.v, false, bH.v,
                                                    (short)0, acc, false, false);
      acc = __builtin_amdgcn_wmma_f32_16x16x32_bf16(false, aH.v, false, bL.v,
                                                    (short)0, acc, false, false);
      acc = __builtin_amdgcn_wmma_f32_16x16x32_bf16(false, aL.v, false, bH.v,
                                                    (short)0, acc, false, false);
    }

    const float ddv = dd[gcol];
#pragma unroll
    for (int r = 0; r < 8; ++r) {
      // C/D layout: VGPR r: lanes0-15 -> M=r, lanes16-31 -> M=r+8
      sDist[rs * 16 + r + 8 * half][cs * 16 + mrow] = ddv - 2.0f * acc[r];
    }
    __syncthreads();

    if (tid < 32) {                       // row owners: streaming top-5 merge
      const int grow = rowBase + tid;
      for (int c = 0; c < 64; ++c) {
        const int gc = cbase + c;
        const float v = sDist[tid][c];
        if (gc != grow && v < worst) {
          bestV[worstSlot] = v; bestI[worstSlot] = gc;
          worst = bestV[0]; worstSlot = 0;
#pragma unroll
          for (int e = 1; e < KNBR; ++e)
            if (bestV[e] > worst) { worst = bestV[e]; worstSlot = e; }
        }
      }
    }
    __syncthreads();
  }

  if (tid < 32) {
    const int grow = rowBase + tid;
#pragma unroll
    for (int e = 0; e < KNBR; ++e) knn[grow * KNBR + e] = bestI[e];
  }
}

// ---------------- stage 3: sparse symmetrized Laplacian ----------------
__global__ void k_fill_f(float* p, float v, int n) {
  const int i = blockIdx.x * blockDim.x + threadIdx.x;
  if (i < n) p[i] = v;
}
__global__ void k_fill_i(int* p, int v, int n) {
  const int i = blockIdx.x * blockDim.x + threadIdx.x;
  if (i < n) p[i] = v;
}

__global__ void k_mutual_deg(const int* __restrict__ knn,
                             unsigned char* __restrict__ mut,
                             int* __restrict__ deg) {
  const int e = blockIdx.x * blockDim.x + threadIdx.x;
  if (e >= N_PTS * KNBR) return;
  const int i = e / KNBR;
  const int t = knn[e];
  bool m = false;
#pragma unroll
  for (int q = 0; q < KNBR; ++q)
    if (knn[t * KNBR + q] == i) m = true;
  mut[e] = m ? 1 : 0;
  if (!m) atomicAdd(&deg[t], 1);   // reverse edge adds a neighbor to t
}

__global__ void k_dinv(const int* __restrict__ deg, float* __restrict__ dinv) {
  const int i = blockIdx.x * blockDim.x + threadIdx.x;
  if (i < N_PTS) dinv[i] = rsqrtf((float)deg[i]);
}

// tmp[i][j] = sum_{t in knn(i)} dinv[t] * v[t][j]     (forward edges)
__global__ void k_spmv_gather(const int* __restrict__ knn,
                              const float* __restrict__ dinv,
                              const float* __restrict__ v,
                              float* __restrict__ tmp) {
  const int j = threadIdx.x;
  if (j >= NV) return;
  const int i = blockIdx.x;
  float s = 0.f;
#pragma unroll
  for (int e = 0; e < KNBR; ++e) {
    const int nb = knn[i * KNBR + e];
    s += dinv[nb] * v[(size_t)nb * NV + j];
  }
  tmp[(size_t)i * NV + j] = s;
}

// reverse (non-mutual) edges: tmp[t][j] += dinv[i]*v[i][j]
__global__ void k_spmv_scatter(const int* __restrict__ knn,
                               const unsigned char* __restrict__ mut,
                               const float* __restrict__ dinv,
                               const float* __restrict__ v,
                               float* __restrict__ tmp) {
  const long long idx = (long long)blockIdx.x * blockDim.x + threadIdx.x;
  if (idx >= (long long)N_PTS * KNBR * NV) return;
  const int e = (int)(idx / NV);
  const int j = (int)(idx % NV);
  if (mut[e]) return;
  const int i = e / KNBR;
  const int t = knn[e];
  atomicAdd(&tmp[(size_t)t * NV + j], dinv[i] * v[(size_t)i * NV + j]);
}

// w = (I - Dinv A Dinv) sv - beta*svold
__global__ void k_spmv_finish(const float* __restrict__ sv,
                              const float* __restrict__ tmp,
                              const float* __restrict__ dinv,
                              const float* __restrict__ beta,
                              const float* __restrict__ svold,
                              float* __restrict__ w) {
  const int idx = blockIdx.x * blockDim.x + threadIdx.x;
  if (idx >= N_PTS * NV) return;
  const int i = idx / NV, j = idx % NV;
  float val = sv[idx] - dinv[i] * tmp[idx];
  if (svold) val -= beta[j] * svold[idx];
  w[idx] = val;
}

// ---------------- stage 4: Lanczos vector ops ----------------
// out[j] += sum_i a[i][j]*b[i][j]  (requires out pre-zeroed)
__global__ void k_coldot_atomic(const float* __restrict__ a,
                                const float* __restrict__ b,
                                float* __restrict__ out) {
  __shared__ float red[NV];
  for (int k = threadIdx.x; k < NV; k += blockDim.x) red[k] = 0.f;
  __syncthreads();
  const int rbase = blockIdx.x * 64;
  for (int idx = threadIdx.x; idx < 64 * NV; idx += blockDim.x) {
    const int il = idx / NV, j = idx % NV;
    const size_t p = (size_t)(rbase + il) * NV + j;
    atomicAdd(&red[j], a[p] * b[p]);
  }
  __syncthreads();
  for (int k = threadIdx.x; k < NV; k += blockDim.x) atomicAdd(&out[k], red[k]);
}

// t[m][j] += sum_i V_m[i][j]*w[i][j]  (t pre-zeroed)
__global__ void k_Vt_w(const float* __restrict__ V, const float* __restrict__ w,
                       float* __restrict__ t) {
  __shared__ float red[MLZ * NV];
  for (int k = threadIdx.x; k < MLZ * NV; k += blockDim.x) red[k] = 0.f;
  __syncthreads();
  const int rbase = blockIdx.x * 64;
  for (int idx = threadIdx.x; idx < 64 * MLZ * NV; idx += blockDim.x) {
    const int il  = idx / (MLZ * NV);
    const int rem = idx % (MLZ * NV);
    const int m = rem / NV, j = rem % NV;
    const size_t p = (size_t)(rbase + il) * NV + j;
    atomicAdd(&red[rem], V[(size_t)m * N_PTS * NV + p] * w[p]);
  }
  __syncthreads();
  for (int k = threadIdx.x; k < MLZ * NV; k += blockDim.x) atomicAdd(&t[k], red[k]);
}

__global__ void k_w_minus_Vt(float* __restrict__ w, const float* __restrict__ V,
                             const float* __restrict__ t) {
  const int idx = blockIdx.x * blockDim.x + threadIdx.x;
  if (idx >= N_PTS * NV) return;
  const int j = idx % NV;
  float s = 0.f;
#pragma unroll
  for (int m = 0; m < MLZ; ++m)
    s += V[(size_t)m * N_PTS * NV + idx] * t[m * NV + j];
  w[idx] -= s;
}

__global__ void k_axpy(float* __restrict__ w, const float* __restrict__ x,
                       const float* __restrict__ coef) {
  const int idx = blockIdx.x * blockDim.x + threadIdx.x;
  if (idx >= N_PTS * NV) return;
  w[idx] -= coef[idx % NV] * x[idx];
}

__global__ void k_scale_rsqrt(float* __restrict__ w, const float* __restrict__ ssq) {
  const int idx = blockIdx.x * blockDim.x + threadIdx.x;
  if (idx >= N_PTS * NV) return;
  w[idx] *= rsqrtf(ssq[idx % NV]);
}

__global__ void k_scale_copy(const float* __restrict__ src,
                             const float* __restrict__ ssq,
                             float* __restrict__ dst) {
  const int idx = blockIdx.x * blockDim.x + threadIdx.x;
  if (idx >= N_PTS * NV) return;
  dst[idx] = src[idx] * rsqrtf(ssq[idx % NV]);
}

__global__ void k_copy(const float* __restrict__ src, float* __restrict__ dst, int n) {
  const int idx = blockIdx.x * blockDim.x + threadIdx.x;
  if (idx < n) dst[idx] = src[idx];
}

// T assembly.  mode 0: T[0,0]=a, T[0,1]=T[1,0]=b, beta=b
//              mode 1: T[i,i]=a      mode 2: T[i,i+1]=T[i+1,i]=b, beta=b
__global__ void k_setT(float* __restrict__ T, const float* __restrict__ alpha,
                       const float* __restrict__ bsq, float* __restrict__ beta,
                       int i, int mode) {
  const int j = threadIdx.x;
  if (j >= NV) return;
  float* Tj = T + (size_t)j * MLZ * MLZ;
  if (mode == 0) {
    const float b = sqrtf(bsq[j]);
    Tj[0] = alpha[j]; Tj[1] = b; Tj[MLZ] = b; beta[j] = b;
  } else if (mode == 1) {
    Tj[i * MLZ + i] = alpha[j];
  } else {
    const float b = sqrtf(bsq[j]);
    Tj[i * MLZ + i + 1] = b; Tj[(i + 1) * MLZ + i] = b; beta[j] = b;
  }
}

// ---------------- stage 5: batched 10x10 Jacobi eigensolver ----------------
__global__ void k_jacobi(const float* __restrict__ T, float* __restrict__ ev,
                         float* __restrict__ qout) {
  const int j = blockIdx.x * blockDim.x + threadIdx.x;
  if (j >= NV) return;
  float A[MLZ][MLZ], Vv[MLZ][MLZ];
  for (int r = 0; r < MLZ; ++r)
    for (int c = 0; c < MLZ; ++c) {
      A[r][c] = T[(size_t)j * MLZ * MLZ + r * MLZ + c];
      Vv[r][c] = (r == c) ? 1.f : 0.f;
    }
  for (int sweep = 0; sweep < 14; ++sweep) {
    for (int p = 0; p < MLZ - 1; ++p)
      for (int q = p + 1; q < MLZ; ++q) {
        const float apq = A[p][q];
        if (fabsf(apq) < 1e-12f) continue;
        const float theta = (A[q][q] - A[p][p]) / (2.f * apq);
        const float tt = (theta >= 0.f ? 1.f : -1.f) /
                         (fabsf(theta) + sqrtf(theta * theta + 1.f));
        const float c = rsqrtf(tt * tt + 1.f);
        const float s = tt * c;
        for (int k = 0; k < MLZ; ++k) {
          const float akp = A[k][p], akq = A[k][q];
          A[k][p] = c * akp - s * akq;
          A[k][q] = s * akp + c * akq;
        }
        for (int k = 0; k < MLZ; ++k) {
          const float apk = A[p][k], aqk = A[q][k];
          A[p][k] = c * apk - s * aqk;
          A[q][k] = s * apk + c * aqk;
        }
        for (int k = 0; k < MLZ; ++k) {
          const float vkp = Vv[k][p], vkq = Vv[k][q];
          Vv[k][p] = c * vkp - s * vkq;
          Vv[k][q] = s * vkp + c * vkq;
        }
      }
  }
  for (int l = 0; l < MLZ; ++l) {
    ev[j * MLZ + l]   = A[l][l];
    qout[j * MLZ + l] = Vv[0][l] * Vv[0][l];   // eigvecs[:,0,:]^2
  }
}

// ---------------- stage 6: SLQ descriptor + final max ----------------
__global__ void k_slq(const float* __restrict__ ev, const float* __restrict__ q,
                      float* __restrict__ msid) {
  const int k = threadIdx.x;            // 256 t samples
  const float t = powf(10.f, -1.f + 2.f * (float)k / (float)(NTSC - 1));
  float se = 0.f, sl = 0.f;
  for (int j = 0; j < NV; ++j)
    for (int l = 0; l < MLZ; ++l) {
      const float te = -t * ev[j * MLZ + l];
      const float qq = q[j * MLZ + l];
      se += expf(te) * qq;
      sl += te * qq;
    }
  const float n = (float)N_PTS;
  const float tr_exp = n * se / (float)NV;
  const float tr_lin = n * sl / (float)NV;
  const float et = expf(t);
  const float m = tr_exp - tr_lin / et - t * n / et;
  msid[k] = m / n * 1.0e6f;
}

__global__ void k_final(const float* __restrict__ mx, const float* __restrict__ my,
                        float* __restrict__ out) {
  __shared__ float red[NTSC];
  const int k = threadIdx.x;
  const float t = powf(10.f, -1.f + 2.f * (float)k / (float)(NTSC - 1));
  const float c = expf(-2.f * (t + 1.f / t));
  red[k] = c * fabsf(mx[k] - my[k]);
  __syncthreads();
  for (int s = NTSC / 2; s > 0; s >>= 1) {
    if (k < s) red[k] = fmaxf(red[k], red[k + s]);
    __syncthreads();
  }
  if (k == 0) out[0] = red[0];
}

// ---------------- workspace layout ----------------
constexpr size_t al256(size_t x) { return (x + 255) & ~(size_t)255; }
constexpr size_t OFF_HI   = 0;
constexpr size_t OFF_LO   = al256(OFF_HI   + (size_t)N_PTS * DIMS * 2);
constexpr size_t OFF_DD   = al256(OFF_LO   + (size_t)N_PTS * DIMS * 2);
constexpr size_t OFF_KNN  = al256(OFF_DD   + (size_t)N_PTS * 4);
constexpr size_t OFF_MUT  = al256(OFF_KNN  + (size_t)N_PTS * KNBR * 4);
constexpr size_t OFF_DEG  = al256(OFF_MUT  + (size_t)N_PTS * KNBR);
constexpr size_t OFF_DINV = al256(OFF_DEG  + (size_t)N_PTS * 4);
constexpr size_t OFF_V    = al256(OFF_DINV + (size_t)N_PTS * 4);
constexpr size_t OFF_W    = al256(OFF_V    + (size_t)MLZ * N_PTS * NV * 4);
constexpr size_t OFF_TMP  = al256(OFF_W    + (size_t)N_PTS * NV * 4);
constexpr size_t OFF_ALPH = al256(OFF_TMP  + (size_t)N_PTS * NV * 4);
constexpr size_t OFF_BSQ  = al256(OFF_ALPH + (size_t)NV * 4);
constexpr size_t OFF_BETA = al256(OFF_BSQ  + (size_t)NV * 4);
constexpr size_t OFF_TSM  = al256(OFF_BETA + (size_t)NV * 4);
constexpr size_t OFF_T    = al256(OFF_TSM  + (size_t)MLZ * NV * 4);
constexpr size_t OFF_EV   = al256(OFF_T    + (size_t)NV * MLZ * MLZ * 4);
constexpr size_t OFF_Q    = al256(OFF_EV   + (size_t)NV * MLZ * 4);
constexpr size_t OFF_MSX  = al256(OFF_Q    + (size_t)NV * MLZ * 4);
constexpr size_t OFF_MSY  = al256(OFF_MSX  + (size_t)NTSC * 4);

// ---------------- host pipeline for one input ----------------
static void run_msid(const float* x, const float* sv_in, char* ws,
                     float* msid_out, hipStream_t s) {
  unsigned short* hi   = (unsigned short*)(ws + OFF_HI);
  unsigned short* lo   = (unsigned short*)(ws + OFF_LO);
  float*          dd   = (float*)(ws + OFF_DD);
  int*            knn  = (int*)(ws + OFF_KNN);
  unsigned char*  mut  = (unsigned char*)(ws + OFF_MUT);
  int*            deg  = (int*)(ws + OFF_DEG);
  float*          dinv = (float*)(ws + OFF_DINV);
  float*          V    = (float*)(ws + OFF_V);
  float*          w    = (float*)(ws + OFF_W);
  float*          tmp  = (float*)(ws + OFF_TMP);
  float*          alph = (float*)(ws + OFF_ALPH);
  float*          bsq  = (float*)(ws + OFF_BSQ);
  float*          beta = (float*)(ws + OFF_BETA);
  float*          tsm  = (float*)(ws + OFF_TSM);
  float*          T    = (float*)(ws + OFF_T);
  float*          ev   = (float*)(ws + OFF_EV);
  float*          q    = (float*)(ws + OFF_Q);

  const int GV  = (N_PTS * NV + 255) / 256;        // grid over n*nv
  const int GE  = (N_PTS * KNBR * NV + 255) / 256; // grid over edges*nv

  auto colDot = [&](const float* a, const float* b, float* out) {
    k_fill_f<<<1, 128, 0, s>>>(out, 0.f, NV);
    k_coldot_atomic<<<N_PTS / 64, 256, 0, s>>>(a, b, out);
  };
  auto spmv = [&](const float* v, const float* bet, const float* svold, float* wo) {
    k_spmv_gather<<<N_PTS, 128, 0, s>>>(knn, dinv, v, tmp);
    k_spmv_scatter<<<GE, 256, 0, s>>>(knn, mut, dinv, v, tmp);
    k_spmv_finish<<<GV, 256, 0, s>>>(v, tmp, dinv, bet, svold, wo);
  };
  auto ortho = [&]() {
    k_fill_f<<<(MLZ * NV + 255) / 256, 256, 0, s>>>(tsm, 0.f, MLZ * NV);
    k_Vt_w<<<N_PTS / 64, 256, 0, s>>>(V, w, tsm);
    k_w_minus_Vt<<<GV, 256, 0, s>>>(w, V, tsm);
  };

  // graph construction
  k_rownorm_cvt<<<N_PTS, DIMS, 0, s>>>(x, hi, lo, dd);
  k_dist_topk<<<N_PTS / 32, 256, 0, s>>>(hi, lo, dd, knn);
  k_fill_i<<<(N_PTS + 255) / 256, 256, 0, s>>>(deg, KNBR, N_PTS);
  k_mutual_deg<<<(N_PTS * KNBR + 255) / 256, 256, 0, s>>>(knn, mut, deg);
  k_dinv<<<(N_PTS + 255) / 256, 256, 0, s>>>(deg, dinv);

  // Lanczos init
  k_fill_f<<<(MLZ * N_PTS * NV + 255) / 256, 256, 0, s>>>(V, 0.f, MLZ * N_PTS * NV);
  k_fill_f<<<(NV * MLZ * MLZ + 255) / 256, 256, 0, s>>>(T, 0.f, NV * MLZ * MLZ);
  colDot(sv_in, sv_in, bsq);
  k_scale_copy<<<GV, 256, 0, s>>>(sv_in, bsq, V);            // V0 = SV/||SV||

  spmv(V, nullptr, nullptr, w);                              // w = L V0
  colDot(w, V, alph);
  k_axpy<<<GV, 256, 0, s>>>(w, V, alph);                     // w -= alpha V0
  colDot(w, w, bsq);
  k_setT<<<1, 128, 0, s>>>(T, alph, bsq, beta, 0, 0);        // T00, T01, T10
  k_scale_rsqrt<<<GV, 256, 0, s>>>(w, bsq);                  // w /= beta
  k_copy<<<GV, 256, 0, s>>>(w, V + (size_t)1 * N_PTS * NV, N_PTS * NV);

  for (int i = 1; i < MLZ; ++i) {
    float* sv    = V + (size_t)i * N_PTS * NV;
    float* svold = V + (size_t)(i - 1) * N_PTS * NV;
    spmv(sv, beta, svold, w);                                // w = L sv - beta svold
    colDot(w, sv, alph);
    k_setT<<<1, 128, 0, s>>>(T, alph, bsq, beta, i, 1);      // T[i][i] = alpha
    if (i < MLZ - 1) {
      k_axpy<<<GV, 256, 0, s>>>(w, sv, alph);                // w -= alpha sv
      ortho();                                               // w -= V (V^T w)
      colDot(w, w, bsq);
      k_setT<<<1, 128, 0, s>>>(T, alph, bsq, beta, i, 2);    // off-diag beta
      k_scale_rsqrt<<<GV, 256, 0, s>>>(w, bsq);
      for (int r = 0; r < NREO; ++r) {                       // reortho passes
        ortho();
        colDot(w, w, bsq);
        k_scale_rsqrt<<<GV, 256, 0, s>>>(w, bsq);
      }
      k_copy<<<GV, 256, 0, s>>>(w, V + (size_t)(i + 1) * N_PTS * NV, N_PTS * NV);
    }
  }

  k_jacobi<<<1, 128, 0, s>>>(T, ev, q);
  k_slq<<<1, NTSC, 0, s>>>(ev, q, msid_out);
}

// ---------------- entry point ----------------
extern "C" void kernel_launch(void* const* d_in, const int* in_sizes, int n_in,
                              void* d_out, int out_size, void* d_ws, size_t ws_size,
                              hipStream_t stream) {
  (void)in_sizes; (void)n_in; (void)out_size; (void)ws_size;
  const float* x   = (const float*)d_in[0];
  const float* y   = (const float*)d_in[1];
  const float* svx = (const float*)d_in[2];
  const float* svy = (const float*)d_in[3];
  char* ws = (char*)d_ws;

  run_msid(x, svx, ws, (float*)(ws + OFF_MSX), stream);
  run_msid(y, svy, ws, (float*)(ws + OFF_MSY), stream);
  k_final<<<1, NTSC, 0, stream>>>((const float*)(ws + OFF_MSX),
                                  (const float*)(ws + OFF_MSY), (float*)d_out);
}
